// CausalSelfAttention_30047591202947
// MI455X (gfx1250) — compile-verified, matrix operand order checked by reading, not run on hardware
//
#include <hip/hip_runtime.h>
#include <hip/hip_bf16.h>

// ---------------------------------------------------------------------------
// CDNA5 (gfx1250) causal self-attention
//   - v_wmma_f32_16x16x32_bf16 for all matmuls (fp32 accumulate)
//   - global_load_async_to_lds_b128 double-buffered staging (ASYNCcnt)
//   - DPP16 row reductions for online softmax (no ds_bpermute)
// ---------------------------------------------------------------------------

typedef __attribute__((ext_vector_type(16))) __bf16 bf16x16;
typedef __attribute__((ext_vector_type(8)))  float  f32x8;

union FragAB { bf16x16 v; unsigned int u[8]; unsigned short s[16]; };
union FragC  { f32x8 v; float f[8]; };

__device__ __forceinline__ unsigned short f2bf(float x) {
    union { float f; unsigned int u; } c; c.f = x;
    unsigned int u = c.u + 0x7fffu + ((c.u >> 16) & 1u);   // round-to-nearest-even
    return (unsigned short)(u >> 16);
}

// K-index base of VGPR pair v for 16-bit A/B fragments (ISA 7.12.2)
__device__ __forceinline__ int kmap(int v, int h2) {
    return ((v >> 2) << 4) + (h2 << 3) + ((v & 3) << 1);
}

// Async direct global->LDS copy, 16 bytes per lane (ISA 10.7 / 15.18.3).
// vdst = LDS byte offset (low 32 bits of flat shared address), vaddr = 32-bit
// byte offset, saddr = uniform 64-bit base.  Tracked with ASYNCcnt.
__device__ __forceinline__ void async_b128(unsigned lds, unsigned goff, const void* base) {
    asm volatile("global_load_async_to_lds_b128 %0, %1, %2"
                 :: "v"(lds), "v"(goff), "s"(base) : "memory");
}
__device__ __forceinline__ unsigned lds_off(const void* p) {
    return (unsigned)(size_t)p;   // flat shared addr low 32 bits == LDS offset
}

// DPP16 cross-lane move confined to each 16-lane row (wave32 halves).
template <int CTRL>
__device__ __forceinline__ float dppf(float x) {
    return __int_as_float(
        __builtin_amdgcn_update_dpp(0, __float_as_int(x), CTRL, 0xF, 0xF, true));
}
// reduce across the 16 lanes of each half: xor1, xor2, half_mirror, mirror
__device__ __forceinline__ float redmax16(float x) {
    x = fmaxf(x, dppf<0xB1>(x));    // quad_perm [1,0,3,2]
    x = fmaxf(x, dppf<0x4E>(x));    // quad_perm [2,3,0,1]
    x = fmaxf(x, dppf<0x141>(x));   // row_half_mirror
    x = fmaxf(x, dppf<0x140>(x));   // row_mirror
    return x;
}
__device__ __forceinline__ float redsum16(float x) {
    x += dppf<0xB1>(x);
    x += dppf<0x4E>(x);
    x += dppf<0x141>(x);
    x += dppf<0x140>(x);
    return x;
}

// ---------------------------------------------------------------------------
// fp32 -> bf16 conversion
// ---------------------------------------------------------------------------
__global__ void cvt_f32_bf16(const float* __restrict__ in,
                             unsigned short* __restrict__ out, int n) {
    int i = blockIdx.x * blockDim.x + threadIdx.x;
    if (i < n) out[i] = f2bf(in[i]);
}

// ---------------------------------------------------------------------------
// Tiled WMMA GEMM: C[M,N] = A[M,K](bf16) * B[K,N](bf16)
// 256 threads (8 waves), tile 64x64, BK=32, double-buffered async staging.
// ---------------------------------------------------------------------------
template <bool OUT_BF16>
__global__ void __launch_bounds__(256)
gemm_wmma(const unsigned short* __restrict__ A,
          const unsigned short* __restrict__ Bm,
          void* __restrict__ Cout, int M, int N, int K) {
    __shared__ __align__(16) unsigned short As[2][64][40];
    __shared__ __align__(16) unsigned short Bs[2][32][72];

    const int t    = threadIdx.x;
    const int lane = t & 31;
    const int wave = t >> 5;
    const int wm   = wave & 3;
    const int wn   = wave >> 2;
    const int h2   = lane >> 4;
    const int l15  = lane & 15;
    const int rowBase = blockIdx.y * 64;
    const int colBase = blockIdx.x * 64;

    FragC acc0, acc1;
#pragma unroll
    for (int i = 0; i < 8; i++) { acc0.f[i] = 0.f; acc1.f[i] = 0.f; }

    const int ar = t >> 2, ac = (t & 3) * 8;   // A stage coords (64x32)
    const int br = t >> 3, bc = (t & 7) * 8;   // B stage coords (32x64)

    const unsigned goffA0 = (unsigned)(((rowBase + ar) * K + ac) * 2);
    const unsigned goffB0 = (unsigned)((br * N + colBase + bc) * 2);
    const int KT = K / 32;

    // prologue: stage tile 0 (2 async instructions per wave)
    async_b128(lds_off(&As[0][ar][ac]), goffA0, A);
    async_b128(lds_off(&Bs[0][br][bc]), goffB0, Bm);

    for (int kt = 0; kt < KT; kt++) {
        const int cur = kt & 1;
        __syncthreads();                      // retire readers of buf cur^1
        if (kt + 1 < KT) {
            async_b128(lds_off(&As[cur ^ 1][ar][ac]), goffA0 + (unsigned)((kt + 1) * 64), A);
            async_b128(lds_off(&Bs[cur ^ 1][br][bc]), goffB0 + (unsigned)((kt + 1) * 32 * N * 2), Bm);
            asm volatile("s_wait_asynccnt 0x2" ::: "memory");
        } else {
            asm volatile("s_wait_asynccnt 0x0" ::: "memory");
        }
        __syncthreads();                      // tile kt resident for all waves

        FragAB a, b0, b1;
        const unsigned short* arow = &As[cur][wm * 16 + l15][0];
#pragma unroll
        for (int v = 0; v < 8; v++)
            a.u[v] = *(const unsigned int*)(arow + kmap(v, h2));
        const int c0 = wn * 32 + l15;
#pragma unroll
        for (int v = 0; v < 8; v++) {
            int k = kmap(v, h2);
            b0.s[2 * v]     = Bs[cur][k][c0];
            b0.s[2 * v + 1] = Bs[cur][k + 1][c0];
            b1.s[2 * v]     = Bs[cur][k][c0 + 16];
            b1.s[2 * v + 1] = Bs[cur][k + 1][c0 + 16];
        }
        acc0.v = __builtin_amdgcn_wmma_f32_16x16x32_bf16(false, a.v, false, b0.v,
                                                         (short)0, acc0.v, false, false);
        acc1.v = __builtin_amdgcn_wmma_f32_16x16x32_bf16(false, a.v, false, b1.v,
                                                         (short)0, acc1.v, false, false);
    }

    const int orow = rowBase + wm * 16;
    const int ocol = colBase + wn * 32 + l15;
#pragma unroll
    for (int i = 0; i < 8; i++) {
        int r = orow + i + 8 * h2;
        if (OUT_BF16) {
            ((unsigned short*)Cout)[(size_t)r * N + ocol]      = f2bf(acc0.f[i]);
            ((unsigned short*)Cout)[(size_t)r * N + ocol + 16] = f2bf(acc1.f[i]);
        } else {
            ((float*)Cout)[(size_t)r * N + ocol]      = acc0.f[i];
            ((float*)Cout)[(size_t)r * N + ocol + 16] = acc1.f[i];
        }
    }
}

// ---------------------------------------------------------------------------
// Flash attention: grid (T/64, H, B), block 128 (4 waves).
// Double-buffered async K/V staging; DPP16 softmax reductions.
// qkv layout: [B*T, 768] bf16 (Q at col h*64, K at 256+h*64, V at 512+h*64).
// ---------------------------------------------------------------------------
__global__ void __launch_bounds__(128)
attn_flash(const unsigned short* __restrict__ qkv, unsigned short* __restrict__ y) {
    constexpr int T = 2048, HS = 64, C3 = 768;
    __shared__ __align__(16) unsigned short Ks[2][32][72];
    __shared__ __align__(16) unsigned short Vs[2][32][72];
    __shared__ __align__(16) unsigned short Ps[4][16][40];

    const int t    = threadIdx.x;
    const int lane = t & 31;
    const int wave = t >> 5;            // 0..3
    const int h2   = lane >> 4;
    const int l15  = lane & 15;
    const int qb   = blockIdx.x * 64;
    const int h    = blockIdx.y;
    const int b    = blockIdx.z;
    const size_t baseQ = (size_t)b * T * C3 + (size_t)h * HS;
    const unsigned short* Kbase = qkv + baseQ + 256;
    const unsigned short* Vbase = qkv + baseQ + 512;

    // Preload this wave's Q fragments (16 x 64 = two K=32 A-fragments)
    FragAB qf[2];
    {
        const int qrow = qb + wave * 16 + l15;
        const unsigned short* qr = qkv + baseQ + (size_t)qrow * C3;
#pragma unroll
        for (int kk = 0; kk < 2; kk++)
#pragma unroll
            for (int v = 0; v < 8; v++)
                qf[kk].u[v] = *(const unsigned int*)(qr + kk * 32 + kmap(v, h2));
    }

    FragC o[4];
#pragma unroll
    for (int d = 0; d < 4; d++)
#pragma unroll
        for (int i = 0; i < 8; i++) o[d].f[i] = 0.f;
    float mrun[8], lrun[8];
#pragma unroll
    for (int i = 0; i < 8; i++) { mrun[i] = -1e30f; lrun[i] = 0.f; }

    const int nkt = qb / 32 + 2;               // causal key-tile bound
    const int lr = t >> 2, lc = (t & 3) * 16;  // cooperative stage coords
    const float scale = 0.125f;                // 1/sqrt(64)

    // per-thread global byte offset of its 32-byte slice within a tile
    const unsigned goff0 = (unsigned)((lr * C3 + lc) * 2);
    const unsigned tileStep = (unsigned)(32 * C3 * 2);

    // prologue: stage tile 0 into buffer 0 (4 async instrs per wave)
    async_b128(lds_off(&Ks[0][lr][lc]),     goff0,      Kbase);
    async_b128(lds_off(&Ks[0][lr][lc + 8]), goff0 + 16, Kbase);
    async_b128(lds_off(&Vs[0][lr][lc]),     goff0,      Vbase);
    async_b128(lds_off(&Vs[0][lr][lc + 8]), goff0 + 16, Vbase);

    for (int j = 0; j < nkt; j++) {
        const int cur = j & 1;
        const int kb  = j * 32;
        __syncthreads();                       // retire readers of buf cur^1 & Ps
        if (j + 1 < nkt) {
            const unsigned g = goff0 + (unsigned)(j + 1) * tileStep;
            async_b128(lds_off(&Ks[cur ^ 1][lr][lc]),     g,      Kbase);
            async_b128(lds_off(&Ks[cur ^ 1][lr][lc + 8]), g + 16, Kbase);
            async_b128(lds_off(&Vs[cur ^ 1][lr][lc]),     g,      Vbase);
            async_b128(lds_off(&Vs[cur ^ 1][lr][lc + 8]), g + 16, Vbase);
            asm volatile("s_wait_asynccnt 0x4" ::: "memory");
        } else {
            asm volatile("s_wait_asynccnt 0x0" ::: "memory");
        }
        __syncthreads();                       // tile j resident for all waves

        // ---- S = Q * K^T (16 x 32, two 16x16 tiles, K-dim = hs = 64) ----
        FragC s[2];
#pragma unroll
        for (int tt = 0; tt < 2; tt++) {
#pragma unroll
            for (int i = 0; i < 8; i++) s[tt].f[i] = 0.f;
#pragma unroll
            for (int kk = 0; kk < 2; kk++) {
                FragAB bK;  // B[d][key]: lane = key col, regs hold d (contig pairs)
                const unsigned short* krow = &Ks[cur][tt * 16 + l15][kk * 32];
#pragma unroll
                for (int v = 0; v < 8; v++)
                    bK.u[v] = *(const unsigned int*)(krow + kmap(v, h2));
                s[tt].v = __builtin_amdgcn_wmma_f32_16x16x32_bf16(
                    false, qf[kk].v, false, bK.v, (short)0, s[tt].v, false, false);
            }
        }

        // ---- causal mask + scale + DPP16 row max ----
        float rm[8];
#pragma unroll
        for (int i = 0; i < 8; i++) {
            const int qr = qb + wave * 16 + i + 8 * h2;
            float v0 = s[0].f[i] * scale, v1 = s[1].f[i] * scale;
            if (kb + l15 > qr)      v0 = -1e30f;
            if (kb + 16 + l15 > qr) v1 = -1e30f;
            s[0].f[i] = v0; s[1].f[i] = v1;
            rm[i] = redmax16(fmaxf(v0, v1));
        }

        // ---- online softmax update ----
        float alpha[8], ls[8];
#pragma unroll
        for (int i = 0; i < 8; i++) {
            float mn = fmaxf(mrun[i], rm[i]);
            alpha[i] = __expf(mrun[i] - mn);
            mrun[i] = mn;
            float p0 = __expf(s[0].f[i] - mn);
            float p1 = __expf(s[1].f[i] - mn);
            s[0].f[i] = p0; s[1].f[i] = p1;
            ls[i] = redsum16(p0 + p1);
        }
#pragma unroll
        for (int i = 0; i < 8; i++) {
            lrun[i] = lrun[i] * alpha[i] + ls[i];
#pragma unroll
            for (int d = 0; d < 4; d++) o[d].f[i] *= alpha[i];
        }

        // ---- C-layout -> A-layout via per-wave LDS, then O += P * V ----
#pragma unroll
        for (int i = 0; i < 8; i++) {
            const int m = i + 8 * h2;
            Ps[wave][m][l15]      = f2bf(s[0].f[i]);
            Ps[wave][m][16 + l15] = f2bf(s[1].f[i]);
        }
        __syncthreads();
        FragAB pf;
        {
            const unsigned short* prow = &Ps[wave][l15][0];
#pragma unroll
            for (int v = 0; v < 8; v++)
                pf.u[v] = *(const unsigned int*)(prow + kmap(v, h2));
        }
#pragma unroll
        for (int d = 0; d < 4; d++) {
            FragAB bV;  // B[key][dcol]: lane = d col, regs hold key
            const int c = d * 16 + l15;
#pragma unroll
            for (int v = 0; v < 8; v++) {
                int k = kmap(v, h2);
                bV.s[2 * v]     = Vs[cur][k][c];
                bV.s[2 * v + 1] = Vs[cur][k + 1][c];
            }
            o[d].v = __builtin_amdgcn_wmma_f32_16x16x32_bf16(
                false, pf.v, false, bV.v, (short)0, o[d].v, false, false);
        }
    }

    // ---- normalize and store y (bf16, layout [B*T, 256]) ----
#pragma unroll
    for (int i = 0; i < 8; i++) {
        const int qr = qb + wave * 16 + i + 8 * h2;
        const float inv = 1.0f / lrun[i];
        unsigned short* yrow = y + ((size_t)b * T + qr) * 256 + h * HS;
#pragma unroll
        for (int d = 0; d < 4; d++)
            yrow[d * 16 + l15] = f2bf(o[d].f[i] * inv);
    }
}

// ---------------------------------------------------------------------------
// launcher
// ---------------------------------------------------------------------------
extern "C" void kernel_launch(void* const* d_in, const int* in_sizes, int n_in,
                              void* d_out, int out_size, void* d_ws, size_t ws_size,
                              hipStream_t stream) {
    const float* x  = (const float*)d_in[0];   // [8,2048,256]
    const float* Wa = (const float*)d_in[1];   // [256,768]
    const float* Wp = (const float*)d_in[2];   // [256,256]

    constexpr int Bsz = 8, T = 2048, C = 256;
    constexpr size_t M = (size_t)Bsz * T;      // 16384

    // workspace layout (bf16 = ushort); y reuses xb (x dead after GEMM1)
    unsigned short* xb   = (unsigned short*)d_ws;        // M*C
    unsigned short* Wab  = xb + M * C;                   // C*3C
    unsigned short* Wpb  = Wab + (size_t)C * 3 * C;      // C*C
    unsigned short* qkvb = Wpb + (size_t)C * C;          // M*3C
    unsigned short* yb   = xb;

    const int nx = (int)(M * C), na = C * 3 * C, np = C * C;
    cvt_f32_bf16<<<(nx + 255) / 256, 256, 0, stream>>>(x, xb, nx);
    cvt_f32_bf16<<<(na + 255) / 256, 256, 0, stream>>>(Wa, Wab, na);
    cvt_f32_bf16<<<(np + 255) / 256, 256, 0, stream>>>(Wp, Wpb, np);

    // qkv = x @ W_attn   [16384,256] x [256,768] -> bf16
    gemm_wmma<true><<<dim3(3 * C / 64, (int)(M / 64)), 256, 0, stream>>>(
        xb, Wab, qkvb, (int)M, 3 * C, C);

    // flash attention -> yb (bf16, [16384,256])
    attn_flash<<<dim3(T / 64, 4, Bsz), 128, 0, stream>>>(qkvb, yb);

    // out = y @ W_proj   [16384,256] x [256,256] -> fp32
    gemm_wmma<false><<<dim3(C / 64, (int)(M / 64)), 256, 0, stream>>>(
        yb, Wpb, d_out, (int)M, C, C);
}
